// TransformerEncoderLayer_30159260352953
// MI455X (gfx1250) — 16-bit and 8-bit matrix operand order repaired and checked by reading, compiled, not run on hardware
//
#include <hip/hip_runtime.h>
#include <hip/hip_bf16.h>
#include <stdint.h>

typedef __attribute__((ext_vector_type(4)))  int       v4i;
typedef __attribute__((ext_vector_type(8)))  int       v8i;
typedef __attribute__((ext_vector_type(8)))  float     v8f;
typedef __attribute__((ext_vector_type(16))) _Float16  v16h;

static constexpr int B_  = 8;
static constexpr int S_  = 512;
static constexpr int D_  = 768;
static constexpr int F_  = 3072;
static constexpr int H_  = 12;
static constexpr int DH_ = 64;
static constexpr int M_  = B_ * S_;
static constexpr size_t BSD = (size_t)M_ * D_;
static constexpr size_t BSF = (size_t)M_ * F_;

__global__ void k_zero_f32(float* p, int n) {
  int i = blockIdx.x * blockDim.x + threadIdx.x;
  if (i < n) p[i] = 0.f;
}

__global__ __launch_bounds__(256) void k_absmax(const float* __restrict__ x,
                                                size_t n, unsigned* slot) {
  __shared__ float red[256];
  float m = 0.f;
  for (size_t i = (size_t)blockIdx.x * 256 + threadIdx.x; i < n;
       i += (size_t)gridDim.x * 256)
    m = fmaxf(m, fabsf(x[i]));
  red[threadIdx.x] = m;
  __syncthreads();
  for (int o = 128; o > 0; o >>= 1) {
    if (threadIdx.x < o) red[threadIdx.x] = fmaxf(red[threadIdx.x], red[threadIdx.x + o]);
    __syncthreads();
  }
  if (threadIdx.x == 0) atomicMax(slot, __float_as_uint(red[0]));
}

__global__ __launch_bounds__(256) void k_tile_absmax(const float* __restrict__ x,
                                                     int D, unsigned* slots) {
  __shared__ float red[256];
  int tile = blockIdx.x >> 3;
  int b    = blockIdx.x & 7;
  size_t cnt = (size_t)32 * D;
  const float* base = x + ((size_t)b * S_ + (size_t)tile * 32) * D;
  float m = 0.f;
  for (size_t i = threadIdx.x; i < cnt; i += 256) m = fmaxf(m, fabsf(base[i]));
  red[threadIdx.x] = m;
  __syncthreads();
  for (int o = 128; o > 0; o >>= 1) {
    if (threadIdx.x < o) red[threadIdx.x] = fmaxf(red[threadIdx.x], red[threadIdx.x + o]);
    __syncthreads();
  }
  if (threadIdx.x == 0) atomicMax(&slots[tile], __float_as_uint(red[0]));
}

__device__ inline float mk_scale(float absmax) {
  return fmaxf(absmax * (1.f / 127.f), 1e-8f);
}

__global__ __launch_bounds__(256) void k_quant_i8(const float* __restrict__ x,
                                                  const float* __restrict__ amax,
                                                  int tileMode, int D,
                                                  int8_t* __restrict__ q, size_t n) {
  for (size_t i = (size_t)blockIdx.x * 256 + threadIdx.x; i < n;
       i += (size_t)gridDim.x * 256) {
    int slot = 0;
    if (tileMode) {
      int s = (int)((i / D) % S_);
      slot = s >> 5;
    }
    float sc = mk_scale(amax[slot]);
    float v = fminf(fmaxf(rintf(x[i] / sc), -127.f), 127.f);
    q[i] = (int8_t)v;
  }
}

__global__ __launch_bounds__(256) void k_quant_f32_inplace(float* __restrict__ x,
                                                           const float* __restrict__ amax,
                                                           size_t n) {
  float sc = mk_scale(amax[0]);
  for (size_t i = (size_t)blockIdx.x * 256 + threadIdx.x; i < n;
       i += (size_t)gridDim.x * 256)
    x[i] = fminf(fmaxf(rintf(x[i] / sc), -127.f), 127.f) * sc;
}

__device__ inline void stage_b128(const int8_t* g, int8_t* lds) {
#if __has_builtin(__builtin_amdgcn_global_load_async_to_lds_b128)
  __builtin_amdgcn_global_load_async_to_lds_b128(
      (__attribute__((address_space(1))) v4i*)(uintptr_t)g,
      (__attribute__((address_space(3))) v4i*)(uintptr_t)lds,
      0, 0);
#else
  *(int4*)lds = *(const int4*)g;
#endif
}
__device__ inline void stage_wait() {
#if __has_builtin(__builtin_amdgcn_s_wait_asynccnt)
  __builtin_amdgcn_s_wait_asynccnt(0);
#endif
}

__global__ __launch_bounds__(256) void k_gemm_i8(
    const int8_t* __restrict__ A, const int8_t* __restrict__ Bw,
    const float* __restrict__ bias,
    const float* __restrict__ a_amax, int a_tile,
    const float* __restrict__ b_amax,
    float* __restrict__ Cout, _Float16* __restrict__ Hout,
    int M, int N, int K) {
  __shared__ int8_t ldsB[2][64 * 64];

  int tid  = threadIdx.x;
  int lane = tid & 31;
  int w    = tid >> 5;
  int mBase = (blockIdx.y * 8 + w) * 16;
  int nBase = blockIdx.x * 64;
  int half = lane >> 4;
  int lr   = lane & 15;
  const int8_t* Arow = A + (size_t)(mBase + lr) * K;

  int bn   = tid >> 2;
  int bseg = (tid & 3) << 4;
  const int8_t* Bg = Bw + (size_t)(nBase + bn) * K + bseg;
  int ldsOff = bn * 64 + bseg;

  v8i acc[4];
#pragma unroll
  for (int t = 0; t < 4; ++t) acc[t] = (v8i){0, 0, 0, 0, 0, 0, 0, 0};

  stage_b128(Bg, &ldsB[0][ldsOff]);

  for (int k0 = 0; k0 < K; k0 += 64) {
    int buf = (k0 >> 6) & 1;
    stage_wait();
    __syncthreads();
    if (k0 + 64 < K) stage_b128(Bg + k0 + 64, &ldsB[buf ^ 1][ldsOff]);

    __builtin_prefetch(Arow + k0 + 128, 0, 0);
    v8i av;
#pragma unroll
    for (int i = 0; i < 8; ++i) {
      int ka = ((i & 4) << 3) + ((i & 2) << 3) + ((i & 1) << 2) + half * 8;
      av[i] = *(const int*)(Arow + k0 + ka);
    }
#pragma unroll
    for (int t = 0; t < 4; ++t) {
      const int8_t* brow = &ldsB[buf][(t * 16 + lr) * 64];
      v8i bv;
#pragma unroll
      for (int i = 0; i < 8; ++i) {
        int kb = ((i & 4) << 3) + ((i & 2) << 3) + ((i & 1) << 2) + half * 8;
        bv[i] = *(const int*)(brow + kb);
      }
      acc[t] = __builtin_amdgcn_wmma_i32_16x16x64_iu8(true, av, true, bv, acc[t],
                                                      false, false);
    }
  }

  float sa;
  if (a_tile) {
    int s0 = mBase % S_;
    sa = mk_scale(a_amax[s0 >> 5]);
  } else {
    sa = mk_scale(a_amax[0]);
  }
  float sb = mk_scale(b_amax[0]);

#pragma unroll
  for (int t = 0; t < 4; ++t) {
    int n = nBase + t * 16 + lr;
    float bn_ = bias ? bias[n] : 0.f;
#pragma unroll
    for (int r = 0; r < 8; ++r) {
      int m = mBase + r + half * 8;
      float val = (float)acc[t][r] * sa * sb + bn_;
      if (Cout) Cout[(size_t)m * N + n] = val;
      if (Hout) {
        int b = m >> 9, s = m & (S_ - 1);
        int hh = n >> 6, d = n & (DH_ - 1);
        Hout[(((size_t)(b * H_ + hh) * S_) + s) * DH_ + d] = (_Float16)val;
      }
    }
  }
}

__global__ __launch_bounds__(64) void k_attn(const _Float16* __restrict__ Q,
                                             const _Float16* __restrict__ K,
                                             const _Float16* __restrict__ V,
                                             float* __restrict__ ctx) {
  __shared__ _Float16 sc[2][16][512];
  __shared__ float rmaxS[2][16];
  __shared__ float rsumS[2][16];
  __shared__ float plaM[12], plaC[12];

  int tid = threadIdx.x, w = tid >> 5, lane = tid & 31;
  if (tid < 12) {
    float xs = -10.f + tid * (10.f / 12.f);
    float xe = xs + 10.f / 12.f;
    float es = expf(xs), ee = expf(xe);
    float m = (ee - es) / (xe - xs);
    plaM[tid] = m;
    plaC[tid] = es - m * xs;
  }
  __syncthreads();

  int gw = blockIdx.x * 2 + w;
  int bh = gw >> 5;
  int qt = gw & 31;
  int half = lane >> 4, lr = lane & 15;
  size_t base = (size_t)bh * S_ * DH_;

  v16h qa0, qa1;
  const _Float16* qrow = Q + base + (size_t)(qt * 16 + lr) * DH_;
#pragma unroll
  for (int p = 0; p < 8; ++p) {
    int kp = ((p & 4) << 2) + half * 8 + ((p & 3) << 1);
    qa0[2 * p] = qrow[kp];          qa0[2 * p + 1] = qrow[kp + 1];
    qa1[2 * p] = qrow[32 + kp];     qa1[2 * p + 1] = qrow[32 + kp + 1];
  }

  float mymax[8];
#pragma unroll
  for (int r = 0; r < 8; ++r) mymax[r] = -1e30f;

  for (int kt = 0; kt < 32; ++kt) {
    v16h kb0, kb1;
    const _Float16* krow = K + base + (size_t)(kt * 16 + lr) * DH_;
#pragma unroll
    for (int j = 0; j < 8; ++j) {
      int kk = ((j & 4) << 2) + half * 8 + ((j & 3) << 1);
      kb0[2 * j] = krow[kk];       kb0[2 * j + 1] = krow[kk + 1];
      kb1[2 * j] = krow[32 + kk];  kb1[2 * j + 1] = krow[32 + kk + 1];
    }
    v8f s = {0.f, 0.f, 0.f, 0.f, 0.f, 0.f, 0.f, 0.f};
    s = __builtin_amdgcn_wmma_f32_16x16x32_f16(false, qa0, false, kb0, (short)0, s, false, false);
    s = __builtin_amdgcn_wmma_f32_16x16x32_f16(false, qa1, false, kb1, (short)0, s, false, false);
#pragma unroll
    for (int r = 0; r < 8; ++r) {
      float v = s[r] * 0.125f;
      sc[w][r + half * 8][kt * 16 + lr] = (_Float16)v;
      mymax[r] = fmaxf(mymax[r], v);
    }
  }
#pragma unroll
  for (int msk = 1; msk < 16; msk <<= 1)
#pragma unroll
    for (int r = 0; r < 8; ++r)
      mymax[r] = fmaxf(mymax[r], __shfl_xor(mymax[r], msk, 32));
  if (lr == 0) {
#pragma unroll
    for (int r = 0; r < 8; ++r) rmaxS[w][r + half * 8] = mymax[r];
  }
  __syncthreads();
  float rm = rmaxS[w][lr];

  v8f a0 = {0.f,0.f,0.f,0.f,0.f,0.f,0.f,0.f}, a1 = a0, a2 = a0, a3 = a0;
  float psum = 0.f;
  for (int kk = 0; kk < 16; ++kk) {
    v16h pa;
#pragma unroll
    for (int p = 0; p < 8; ++p) {
      int kp = kk * 32 + ((p & 4) << 2) + half * 8 + ((p & 3) << 1);
#pragma unroll
      for (int e = 0; e < 2; ++e) {
        float x = (float)sc[w][lr][kp + e] - rm;
        x = fminf(fmaxf(x, -10.f), 0.f);
        int idx = (int)floorf((x + 10.f) * 1.2f);
        idx = min(max(idx, 0), 11);
        float ev = plaM[idx] * x + plaC[idx];
        pa[2 * p + e] = (_Float16)ev;
        psum += ev;
      }
    }
    v16h vb0, vb1, vb2, vb3;
#pragma unroll
    for (int j = 0; j < 8; ++j) {
      int ks = kk * 32 + ((j & 4) << 2) + half * 8 + ((j & 3) << 1);
      const _Float16* vrow = V + base + (size_t)ks * DH_;
      vb0[2 * j] = vrow[lr];       vb0[2 * j + 1] = vrow[DH_ + lr];
      vb1[2 * j] = vrow[16 + lr];  vb1[2 * j + 1] = vrow[DH_ + 16 + lr];
      vb2[2 * j] = vrow[32 + lr];  vb2[2 * j + 1] = vrow[DH_ + 32 + lr];
      vb3[2 * j] = vrow[48 + lr];  vb3[2 * j + 1] = vrow[DH_ + 48 + lr];
    }
    a0 = __builtin_amdgcn_wmma_f32_16x16x32_f16(false, pa, false, vb0, (short)0, a0, false, false);
    a1 = __builtin_amdgcn_wmma_f32_16x16x32_f16(false, pa, false, vb1, (short)0, a1, false, false);
    a2 = __builtin_amdgcn_wmma_f32_16x16x32_f16(false, pa, false, vb2, (short)0, a2, false, false);
    a3 = __builtin_amdgcn_wmma_f32_16x16x32_f16(false, pa, false, vb3, (short)0, a3, false, false);
  }
  psum += __shfl_xor(psum, 16, 32);
  rsumS[w][lr] = psum;
  __syncthreads();

  int b = bh / H_, hh = bh % H_;
#pragma unroll
  for (int r = 0; r < 8; ++r) {
    int row = r + half * 8;
    float inv = 1.f / rsumS[w][row];
    int s = qt * 16 + row;
    float* orow = ctx + ((size_t)(b * S_ + s)) * D_ + hh * DH_;
    orow[lr]      = a0[r] * inv;
    orow[16 + lr] = a1[r] * inv;
    orow[32 + lr] = a2[r] * inv;
    orow[48 + lr] = a3[r] * inv;
  }
}

__global__ __launch_bounds__(256) void k_add_ln(const float* __restrict__ A,
                                                const float* __restrict__ Bv,
                                                const float* __restrict__ w,
                                                const float* __restrict__ bias,
                                                const float* __restrict__ qamax,
                                                int quantB,
                                                float* __restrict__ out) {
  __shared__ float red[256], red2[256];
  int row = blockIdx.x, tid = threadIdx.x;
  const float* a = A  + (size_t)row * D_;
  const float* b = Bv + (size_t)row * D_;
  float sc = quantB ? mk_scale(qamax[0]) : 1.f;
  float t[3];
  float sum = 0.f, sq = 0.f;
#pragma unroll
  for (int i = 0; i < 3; ++i) {
    int c = tid + i * 256;
    float bv = b[c];
    if (quantB) bv = fminf(fmaxf(rintf(bv / sc), -127.f), 127.f) * sc;
    float v = a[c] + bv;
    t[i] = v; sum += v; sq += v * v;
  }
  red[tid] = sum; red2[tid] = sq;
  __syncthreads();
  for (int o = 128; o > 0; o >>= 1) {
    if (tid < o) { red[tid] += red[tid + o]; red2[tid] += red2[tid + o]; }
    __syncthreads();
  }
  float mean = red[0] / (float)D_;
  float var  = fmaxf(red2[0] / (float)D_ - mean * mean, 0.f);
  float vf = fminf(fmaxf(rintf(var * 67108864.f), -2147483648.f), 2147483647.f)
             * (1.f / 67108864.f);
  float x0 = vf > 1.f ? vf * 0.5f : 1.f;
#pragma unroll
  for (int it = 0; it < 8; ++it) x0 = 0.5f * (x0 + vf / (x0 + 1e-9f));
  float istd = 1.f / (x0 + 1e-5f);
#pragma unroll
  for (int i = 0; i < 3; ++i) {
    int c = tid + i * 256;
    out[(size_t)row * D_ + c] = w[c] * ((t[i] - mean) * istd) + bias[c];
  }
}

__device__ inline float eu_f(float x, bool log2e) {
  if (log2e) x *= 1.4375f;
  float xi = floorf(x);
  float xf = fminf(fmaxf(x - xi, 0.f), 0.999999f);
  int seg = (int)floorf(xf * 8.f);
  seg = min(max(seg, 0), 7);
  float xs = seg * 0.125f;
  float ys = exp2f(xs), ye = exp2f(xs + 0.125f);
  float Kc = (ye - ys) * 8.f;
  float Bc = ys - Kc * xs;
  return (Kc * xf + Bc) * exp2f(fminf(fmaxf(xi, -15.f), 15.f));
}
__device__ inline float sgn_f(float x) { return (x > 0.f) ? 1.f : ((x < 0.f) ? -1.f : 0.f); }
__device__ inline float du_f(float num, float den, bool add_one) {
  if (add_one) den = 1.f + den;
  float l1 = log2f(fmaxf(fabsf(num), 1e-8f));
  float w1 = floorf(l1), m1 = l1 - w1 + 1.f;
  float l2 = log2f(fmaxf(fabsf(den), 1e-8f));
  float w2 = floorf(l2), m2 = l2 - w2 + 1.f;
  return eu_f(m1 + w1 - (m2 + w2), false) * sgn_f(num) * sgn_f(den);
}
__device__ inline float gcu_f(float x) {
  float sx = -10.3125f * (0.8f * x + 0.0625f * x * x * x);
  return du_f(x, eu_f(-sx, false), true);
}

__global__ __launch_bounds__(256) void k_gcu_absmax(float* __restrict__ x, size_t n,
                                                    unsigned* slot) {
  __shared__ float red[256];
  float m = 0.f;
  for (size_t i = (size_t)blockIdx.x * 256 + threadIdx.x; i < n;
       i += (size_t)gridDim.x * 256) {
    float g = gcu_f(x[i]);
    x[i] = g;
    m = fmaxf(m, fabsf(g));
  }
  red[threadIdx.x] = m;
  __syncthreads();
  for (int o = 128; o > 0; o >>= 1) {
    if (threadIdx.x < o) red[threadIdx.x] = fmaxf(red[threadIdx.x], red[threadIdx.x + o]);
    __syncthreads();
  }
  if (threadIdx.x == 0) atomicMax(slot, __float_as_uint(red[0]));
}

extern "C" void kernel_launch(void* const* d_in, const int* in_sizes, int n_in,
                              void* d_out, int out_size, void* d_ws, size_t ws_size,
                              hipStream_t stream) {
  const float* x    = (const float*)d_in[0];
  const float* wq   = (const float*)d_in[1];
  const float* bq   = (const float*)d_in[2];
  const float* wk   = (const float*)d_in[3];
  const float* bk   = (const float*)d_in[4];
  const float* wv   = (const float*)d_in[5];
  const float* bv   = (const float*)d_in[6];
  const float* wo   = (const float*)d_in[7];
  const float* bo   = (const float*)d_in[8];
  const float* ln1w = (const float*)d_in[9];
  const float* ln1b = (const float*)d_in[10];
  const float* w1   = (const float*)d_in[11];
  const float* b1   = (const float*)d_in[12];
  const float* w2   = (const float*)d_in[13];
  const float* b2   = (const float*)d_in[14];
  const float* ln2w = (const float*)d_in[15];
  const float* ln2b = (const float*)d_in[16];
  float* out = (float*)d_out;

  char* W = (char*)d_ws;
  size_t off = 0;
  auto alloc = [&](size_t bytes) -> void* {
    void* p = W + off;
    off = (off + bytes + 255) & ~(size_t)255;
    return p;
  };
  float*    scales = (float*)alloc(256 * sizeof(float));
  int8_t*   xq   = (int8_t*)alloc(BSD);
  int8_t*   wqq  = (int8_t*)alloc((size_t)D_ * D_);
  int8_t*   wkq  = (int8_t*)alloc((size_t)D_ * D_);
  int8_t*   wvq  = (int8_t*)alloc((size_t)D_ * D_);
  int8_t*   woq  = (int8_t*)alloc((size_t)D_ * D_);
  int8_t*   w1q  = (int8_t*)alloc((size_t)F_ * D_);
  int8_t*   w2q  = (int8_t*)alloc((size_t)D_ * F_);
  _Float16* qh   = (_Float16*)alloc(BSD * 2);
  _Float16* kh   = (_Float16*)alloc(BSD * 2);
  _Float16* vh   = (_Float16*)alloc(BSD * 2);
  float*    ctx  = (float*)alloc(BSD * 4);
  int8_t*   ctxq = (int8_t*)alloc(BSD);
  float*    attn = (float*)alloc(BSD * 4);
  float*    hbuf = (float*)alloc(BSD * 4);
  int8_t*   hq   = (int8_t*)alloc(BSD);
  float*    h1   = (float*)alloc(BSF * 4);
  int8_t*   gq   = (int8_t*)alloc(BSF);
  float*    obuf = (float*)alloc(BSD * 4);
  if (off > ws_size) return;

  unsigned* uscales = (unsigned*)scales;

  k_zero_f32<<<1, 256, 0, stream>>>(scales, 64);

  k_absmax<<<512, 256, 0, stream>>>(x, BSD, uscales + 0);
  k_absmax<<<256, 256, 0, stream>>>(wq, (size_t)D_ * D_, uscales + 1);
  k_absmax<<<256, 256, 0, stream>>>(wk, (size_t)D_ * D_, uscales + 2);
  k_absmax<<<256, 256, 0, stream>>>(wv, (size_t)D_ * D_, uscales + 3);
  k_absmax<<<256, 256, 0, stream>>>(wo, (size_t)D_ * D_, uscales + 4);
  k_absmax<<<512, 256, 0, stream>>>(w1, (size_t)F_ * D_, uscales + 5);
  k_absmax<<<512, 256, 0, stream>>>(w2, (size_t)D_ * F_, uscales + 6);

  k_quant_i8<<<1024, 256, 0, stream>>>(x,  scales + 0, 0, D_, xq,  BSD);
  k_quant_i8<<<512,  256, 0, stream>>>(wq, scales + 1, 0, D_, wqq, (size_t)D_ * D_);
  k_quant_i8<<<512,  256, 0, stream>>>(wk, scales + 2, 0, D_, wkq, (size_t)D_ * D_);
  k_quant_i8<<<512,  256, 0, stream>>>(wv, scales + 3, 0, D_, wvq, (size_t)D_ * D_);
  k_quant_i8<<<512,  256, 0, stream>>>(wo, scales + 4, 0, D_, woq, (size_t)D_ * D_);
  k_quant_i8<<<1024, 256, 0, stream>>>(w1, scales + 5, 0, D_, w1q, (size_t)F_ * D_);
  k_quant_i8<<<1024, 256, 0, stream>>>(w2, scales + 6, 0, F_, w2q, (size_t)D_ * F_);

  dim3 gQKV(D_ / 64, M_ / 128);
  k_gemm_i8<<<gQKV, 256, 0, stream>>>(xq, wqq, bq, scales + 0, 0, scales + 1,
                                      nullptr, qh, M_, D_, D_);
  k_gemm_i8<<<gQKV, 256, 0, stream>>>(xq, wkq, bk, scales + 0, 0, scales + 2,
                                      nullptr, kh, M_, D_, D_);
  k_gemm_i8<<<gQKV, 256, 0, stream>>>(xq, wvq, bv, scales + 0, 0, scales + 3,
                                      nullptr, vh, M_, D_, D_);

  k_attn<<<(B_ * H_ * (S_ / 16)) / 2, 64, 0, stream>>>(qh, kh, vh, ctx);

  k_absmax<<<512, 256, 0, stream>>>(ctx, BSD, uscales + 7);
  k_quant_i8<<<1024, 256, 0, stream>>>(ctx, scales + 7, 0, D_, ctxq, BSD);
  k_gemm_i8<<<gQKV, 256, 0, stream>>>(ctxq, woq, bo, scales + 7, 0, scales + 4,
                                      attn, nullptr, M_, D_, D_);

  k_add_ln<<<M_, 256, 0, stream>>>(x, attn, ln1w, ln1b, nullptr, 0, hbuf);

  k_tile_absmax<<<128, 256, 0, stream>>>(hbuf, D_, uscales + 16);
  k_quant_i8<<<1024, 256, 0, stream>>>(hbuf, scales + 16, 1, D_, hq, BSD);
  dim3 gF1(F_ / 64, M_ / 128);
  k_gemm_i8<<<gF1, 256, 0, stream>>>(hq, w1q, b1, scales + 16, 1, scales + 5,
                                     h1, nullptr, M_, F_, D_);

  k_gcu_absmax<<<2048, 256, 0, stream>>>(h1, BSF, uscales + 8);
  k_quant_f32_inplace<<<2048, 256, 0, stream>>>(h1, scales + 8, BSF);
  k_tile_absmax<<<128, 256, 0, stream>>>(h1, F_, uscales + 32);
  k_quant_i8<<<2048, 256, 0, stream>>>(h1, scales + 32, 1, F_, gq, BSF);

  dim3 gF2(D_ / 64, M_ / 128);
  k_gemm_i8<<<gF2, 256, 0, stream>>>(gq, w2q, b2, scales + 32, 1, scales + 6,
                                     obuf, nullptr, M_, D_, F_);

  k_absmax<<<512, 256, 0, stream>>>(obuf, BSD, uscales + 9);
  k_add_ln<<<M_, 256, 0, stream>>>(hbuf, obuf, ln2w, ln2b, scales + 9, 1, out);
}
